// MultiHeadAttention_77764677862146
// MI455X (gfx1250) — compile-verified, matrix operand order checked by reading, not run on hardware
//
#include <hip/hip_runtime.h>

#define D_MODELC 1024
#define N_HEADC  16
#define D_QKVC   64
#define BATCHC   2
#define SEQC     2048

typedef __attribute__((ext_vector_type(16))) __bf16 v16bf;
typedef __attribute__((ext_vector_type(8)))  float  v8f;
typedef __attribute__((ext_vector_type(4)))  unsigned int u32x4;
typedef __attribute__((ext_vector_type(8)))  int i32x8;
typedef __attribute__((ext_vector_type(4)))  int i32x4;

union Frag {
  v16bf bf;
  uint4 q[2];
};

__device__ __forceinline__ unsigned short f2bf(float f) {
  unsigned int u = __float_as_uint(f);
  u += 0x7FFFu + ((u >> 16) & 1u);          // round-to-nearest-even
  return (unsigned short)(u >> 16);
}

// Load one bf16 WMMA fragment (16x32 A or 32x16 B): per-lane two contiguous
// 16-byte runs at K offsets [8*hf, 8*hf+8) and [16+8*hf, 24+8*hf).
__device__ __forceinline__ Frag load_frag(const unsigned short* p, int hf) {
  Frag f;
  f.q[0] = *(const uint4*)(p + 8 * hf);
  f.q[1] = *(const uint4*)(p + 16 + 8 * hf);
  return f;
}

// Escape an LDS pointer into an asm with a memory clobber so the compiler
// must assume the TDM intrinsic wrote the shared array (otherwise loads from
// a never-stored __shared__ object get folded to undef and DCE'd -- observed
// in round-2 disasm). Also pins the ds_loads below the tensorcnt wait/barrier.
__device__ __forceinline__ void lds_escape(const void* p) {
  asm volatile("" : : "v"(p) : "memory");
}

// ---------------- Tensor Data Mover staging ----------------
#if defined(__has_builtin)
#if __has_builtin(__builtin_amdgcn_tensor_load_to_lds)
#define HAVE_TDM 1
#endif
#endif

#ifdef HAVE_TDM
// 2-D tile load: tile1 rows x tile0 elems (bf16), row stride `stride` elems,
// into LDS at byte offset lds_byte_off. D# per CDNA5 ISA ch.8:
//   g0: count=1 | lds_addr | global_addr | type=2
//   g1: data_size=1(2B) | tensor_dim0=stride | tensor_dim1=tile1 |
//       tile_dim0/1 | tensor_dim0_stride=stride
__device__ __forceinline__ void tdm_stage(unsigned lds_byte_off,
                                          const unsigned short* gsrc,
                                          int tile0, int tile1, int stride) {
  unsigned long long ga = (unsigned long long)(uintptr_t)gsrc;
  u32x4 g0 = { 1u, lds_byte_off, (unsigned)ga,
               (unsigned)(ga >> 32) | 0x80000000u /* type=2 */ };
  i32x8 g1 = { 0x10000 /* data_size=2B */,
               (int)(((unsigned)stride & 0xFFFFu) << 16),
               (int)(((unsigned)stride >> 16) | (((unsigned)tile1 & 0xFFFFu) << 16)),
               (int)(((unsigned)tile0 & 0xFFFFu) << 16),
               tile1, stride, 0, 0 };
  i32x4 z4 = { 0, 0, 0, 0 };
#if __clang_major__ >= 23
  i32x8 z8 = { 0, 0, 0, 0, 0, 0, 0, 0 };
  __builtin_amdgcn_tensor_load_to_lds(g0, g1, z4, z4, z8, 0);
#else
  __builtin_amdgcn_tensor_load_to_lds(g0, g1, z4, z4, 0);
#endif
}
#endif

// Issue a tile stage (one wave calls this). TDM path is async (TENSORcnt).
__device__ __forceinline__ void stage_tile(unsigned short* smem_base,
                                           unsigned lds_byte_off,
                                           const unsigned short* gsrc,
                                           int tile0, int tile1, int stride) {
  (void)smem_base;
#ifdef HAVE_TDM
  tdm_stage(lds_byte_off, gsrc, tile0, tile1, stride);
#else
  const int lane = threadIdx.x & 31;
  unsigned short* dst = smem_base + lds_byte_off / 2;
  const int chunks = (tile0 * tile1) / 8;   // 16B chunks
  for (int c = lane; c < chunks; c += 32) {
    const int row = (c * 8) / tile0;
    const int col = (c * 8) % tile0;
    *(uint4*)(dst + row * tile0 + col) =
        *(const uint4*)(gsrc + (size_t)row * stride + col);
  }
#endif
}

__device__ __forceinline__ void stage_wait() {
#ifdef HAVE_TDM
  __builtin_amdgcn_s_wait_tensorcnt(0);
#endif
}

// ---------------- precision conversion kernels ----------------

__global__ void k_cast_bf16(const float* __restrict__ in,
                            unsigned short* __restrict__ out, int n) {
  for (int i = blockIdx.x * blockDim.x + threadIdx.x; i < n;
       i += gridDim.x * blockDim.x)
    out[i] = f2bf(in[i]);
}

// [H][D][K] fp32 -> [H][K][D] bf16 (so GEMM B-fragments are contiguous in D)
__global__ void k_cast_wT(const float* __restrict__ in,
                          unsigned short* __restrict__ out) {
  int idx = blockIdx.x * blockDim.x + threadIdx.x;
  const int total = N_HEADC * D_MODELC * D_QKVC;
  if (idx >= total) return;
  int k = idx % D_QKVC;
  int d = (idx / D_QKVC) % D_MODELC;
  int h = idx / (D_QKVC * D_MODELC);
  out[((size_t)h * D_QKVC + k) * D_MODELC + d] = f2bf(in[idx]);
}

// ---------------- QKV projection: [B*S,1024] x [1024,64] per head ----------------
// grid (B*S/128, H, 3), block 256 (8 waves). Wave computes 16 rows x 64 cols.
// Weight tile (64 n-rows x 32 k) staged once per block via TDM, double-buffered.
__global__ void __launch_bounds__(256) k_qkv(
    const unsigned short* __restrict__ tgt_bf, const unsigned short* __restrict__ src_bf,
    const unsigned short* __restrict__ WqT, const unsigned short* __restrict__ WkT,
    const unsigned short* __restrict__ WvT,
    const float* __restrict__ bq, const float* __restrict__ bk,
    const float* __restrict__ bv,
    unsigned short* __restrict__ qo, unsigned short* __restrict__ ko,
    unsigned short* __restrict__ vto) {
  __shared__ __align__(16) unsigned short wtile[2 * 2048];  // 2 x 4KB

  const int which = blockIdx.z;
  const unsigned short* act = (which == 0) ? tgt_bf : src_bf;   // q<-tgt, k/v<-src
  const unsigned short* W   = (which == 0) ? WqT : (which == 1) ? WkT : WvT;
  const float* bias         = (which == 0) ? bq  : (which == 1) ? bk  : bv;

  const int h    = blockIdx.y;
  const int wave = threadIdx.x >> 5;
  const int lane = threadIdx.x & 31;
  const int hf   = lane >> 4;
  const int l16  = lane & 15;

  const int row0 = blockIdx.x * 128 + wave * 16;   // global row in [0, B*S)
  const unsigned short* arow = act + (size_t)(row0 + l16) * D_MODELC;
  const unsigned short* wh   = W + (size_t)h * D_QKVC * D_MODELC;

  if (threadIdx.x < 32)
    stage_tile(wtile, 0, wh, 32, 64, D_MODELC);

  v8f acc[4] = {};
  for (int k0 = 0; k0 < D_MODELC; k0 += 32) {
    const int buf = (k0 >> 5) & 1;
    if (threadIdx.x < 32) stage_wait();
    __syncthreads();                       // tile[buf] ready; prior reads done
    lds_escape(wtile);
    if (threadIdx.x < 32 && k0 + 32 < D_MODELC)
      stage_tile(wtile, (unsigned)((buf ^ 1) * 4096), wh + k0 + 32, 32, 64, D_MODELC);

    Frag a = load_frag(arow + k0, hf);
    const unsigned short* wt = wtile + buf * 2048;
#pragma unroll
    for (int nt = 0; nt < 4; ++nt) {
      Frag bw = load_frag(wt + (nt * 16 + l16) * 32, hf);
      acc[nt] = __builtin_amdgcn_wmma_f32_16x16x32_bf16(
          false, a.bf, false, bw.bf, (short)0, acc[nt], false, false);
    }
  }

#pragma unroll
  for (int nt = 0; nt < 4; ++nt) {
    const int n = nt * 16 + l16;
    const float bi = bias[h * D_QKVC + n];
#pragma unroll
    for (int r = 0; r < 8; ++r) {
      const int row = row0 + r + 8 * hf;
      const int bb = row / SEQC, ss = row % SEQC;
      const unsigned short val = f2bf(acc[nt][r] + bi);
      if (which == 0)
        qo[((size_t)(bb * N_HEADC + h) * SEQC + ss) * D_QKVC + n] = val;
      else if (which == 1)
        ko[((size_t)(bb * N_HEADC + h) * SEQC + ss) * D_QKVC + n] = val;
      else  // V stored transposed [b][h][n][s] so PV B-fragments are contiguous
        vto[((size_t)(bb * N_HEADC + h) * D_QKVC + n) * SEQC + ss] = val;
    }
  }
}

// ---------------- flash attention ----------------
// grid (S/128, H, B), block 256 (8 waves, one 16-row query tile each).
// K chunk (32x64, contiguous) and V chunk (64 n-rows x 32 s) staged per block
// via TDM, double-buffered; all 8 waves share them from LDS.
__global__ void __launch_bounds__(256) k_attn(
    const unsigned short* __restrict__ q, const unsigned short* __restrict__ k,
    const unsigned short* __restrict__ vT, const float* __restrict__ mask,
    unsigned short* __restrict__ ctx) {
  // [0,8KB): K bufs; [8KB,16KB): V bufs; [16KB,24KB): P staging (1KB/wave)
  __shared__ __align__(16) unsigned short smem[12288];

  const int b    = blockIdx.z, h = blockIdx.y;
  const int wave = threadIdx.x >> 5;
  const int lane = threadIdx.x & 31;
  const int hf   = lane >> 4;
  const int l16  = lane & 15;
  const int t0   = blockIdx.x * 128 + wave * 16;

  const size_t head = (size_t)(b * N_HEADC + h);
  const unsigned short* qh = q  + head * SEQC * D_QKVC;
  const unsigned short* kh = k  + head * SEQC * D_QKVC;
  const unsigned short* vh = vT + head * D_QKVC * SEQC;
  unsigned short* ps = smem + 8192 + wave * 512;

  Frag aq[2];
#pragma unroll
  for (int kk = 0; kk < 2; ++kk)
    aq[kk] = load_frag(qh + (size_t)(t0 + l16) * D_QKVC + kk * 32, hf);

  float mrow[8], lrow[8];
#pragma unroll
  for (int r = 0; r < 8; ++r) { mrow[r] = -1e30f; lrow[r] = 0.0f; }
  v8f oacc[4] = {};
  const float scale = 0.125f;  // 1/sqrt(64)

  if (threadIdx.x < 32) {
    stage_tile(smem, 0u,    kh, 64, 32, D_QKVC);   // K rows s0..s0+31
    stage_tile(smem, 8192u, vh, 32, 64, SEQC);     // V^T rows n=0..63
  }

  for (int s0 = 0; s0 < SEQC; s0 += 32) {
    const int buf = (s0 >> 5) & 1;
    if (threadIdx.x < 32) stage_wait();
    __syncthreads();
    lds_escape(smem);
    if (threadIdx.x < 32 && s0 + 32 < SEQC) {
      stage_tile(smem, (unsigned)((buf ^ 1) * 4096),
                 kh + (size_t)(s0 + 32) * D_QKVC, 64, 32, D_QKVC);
      stage_tile(smem, (unsigned)(8192 + (buf ^ 1) * 4096),
                 vh + s0 + 32, 32, 64, SEQC);
    }
    const unsigned short* kt = smem + buf * 2048;
    const unsigned short* vt = smem + 4096 + buf * 2048;

    // scores tile 16x32: 2 N-subtiles x 2 K-steps
    v8f sc[2] = {};
#pragma unroll
    for (int nt = 0; nt < 2; ++nt) {
      const unsigned short* kb = kt + (nt * 16 + l16) * D_QKVC;
#pragma unroll
      for (int kk = 0; kk < 2; ++kk) {
        Frag bk_ = load_frag(kb + kk * 32, hf);
        sc[nt] = __builtin_amdgcn_wmma_f32_16x16x32_bf16(
            false, aq[kk].bf, false, bk_.bf, (short)0, sc[nt], false, false);
      }
    }

    // online softmax: mask added BEFORE scaling (matches reference)
#pragma unroll
    for (int r = 0; r < 8; ++r) {
      const int t = t0 + r + 8 * hf;
      float e0 = (sc[0][r] + mask[(size_t)t * SEQC + s0 + l16]) * scale;
      float e1 = (sc[1][r] + mask[(size_t)t * SEQC + s0 + 16 + l16]) * scale;
      float mx = fmaxf(e0, e1);
#pragma unroll
      for (int off = 1; off < 16; off <<= 1)
        mx = fmaxf(mx, __shfl_xor(mx, off, 32));
      const float mnew = fmaxf(mrow[r], mx);
      const float p0 = __expf(e0 - mnew);
      const float p1 = __expf(e1 - mnew);
      float psum = p0 + p1;
#pragma unroll
      for (int off = 1; off < 16; off <<= 1)
        psum += __shfl_xor(psum, off, 32);
      const float corr = __expf(mrow[r] - mnew);
      lrow[r] = lrow[r] * corr + psum;
      mrow[r] = mnew;
#pragma unroll
      for (int j = 0; j < 4; ++j) oacc[j][r] *= corr;
      // stage P (C-layout -> row-major 16x32); per-wave region, and LDS ops
      // of one wave are architecturally in-order -> no barrier needed
      ps[(r + 8 * hf) * 32 + l16]      = f2bf(p0);
      ps[(r + 8 * hf) * 32 + 16 + l16] = f2bf(p1);
    }

    Frag pf;
    pf.q[0] = *(const uint4*)(ps + (size_t)l16 * 32 + 8 * hf);
    pf.q[1] = *(const uint4*)(ps + (size_t)l16 * 32 + 16 + 8 * hf);
#pragma unroll
    for (int j = 0; j < 4; ++j) {
      Frag bv_ = load_frag(vt + (j * 16 + l16) * 32, hf);
      oacc[j] = __builtin_amdgcn_wmma_f32_16x16x32_bf16(
          false, pf.bf, false, bv_.bf, (short)0, oacc[j], false, false);
    }
  }

#pragma unroll
  for (int r = 0; r < 8; ++r) {
    const float inv = 1.0f / lrow[r];
    const int t = t0 + r + 8 * hf;
#pragma unroll
    for (int j = 0; j < 4; ++j)
      ctx[((size_t)(b * SEQC + t)) * D_MODELC + h * D_QKVC + j * 16 + l16] =
          f2bf(oacc[j][r] * inv);
  }
}

// ---------------- output projection: ctx [B*S,1024] x Wo^T + bo ----------------
// Wo row-major [e][d] == B-fragment contiguous-in-K layout. Wo tile (64 e-rows
// x 32 d) staged per block via TDM, double-buffered.
__global__ void __launch_bounds__(256) k_oproj(
    const unsigned short* __restrict__ ctx, const unsigned short* __restrict__ wo,
    const float* __restrict__ bo, float* __restrict__ out) {
  __shared__ __align__(16) unsigned short wtile[2 * 2048];

  const int wave = threadIdx.x >> 5;
  const int lane = threadIdx.x & 31;
  const int hf   = lane >> 4;
  const int l16  = lane & 15;
  const int row0 = blockIdx.x * 128 + wave * 16;
  const int e0   = blockIdx.y * 64;

  const unsigned short* arow = ctx + (size_t)(row0 + l16) * D_MODELC;
  const unsigned short* wrows = wo + (size_t)e0 * D_MODELC;

  if (threadIdx.x < 32)
    stage_tile(wtile, 0, wrows, 32, 64, D_MODELC);

  v8f acc[4] = {};
  for (int k0 = 0; k0 < D_MODELC; k0 += 32) {
    const int buf = (k0 >> 5) & 1;
    if (threadIdx.x < 32) stage_wait();
    __syncthreads();
    lds_escape(wtile);
    if (threadIdx.x < 32 && k0 + 32 < D_MODELC)
      stage_tile(wtile, (unsigned)((buf ^ 1) * 4096), wrows + k0 + 32, 32, 64, D_MODELC);

    Frag a = load_frag(arow + k0, hf);
    const unsigned short* wt = wtile + buf * 2048;
#pragma unroll
    for (int nt = 0; nt < 4; ++nt) {
      Frag bw = load_frag(wt + (nt * 16 + l16) * 32, hf);
      acc[nt] = __builtin_amdgcn_wmma_f32_16x16x32_bf16(
          false, a.bf, false, bw.bf, (short)0, acc[nt], false, false);
    }
  }
#pragma unroll
  for (int nt = 0; nt < 4; ++nt) {
    const int e = e0 + nt * 16 + l16;
    const float bi = bo[e];
#pragma unroll
    for (int r = 0; r < 8; ++r)
      out[(size_t)(row0 + r + 8 * hf) * D_MODELC + e] = acc[nt][r] + bi;
  }
}

// ---------------- launch ----------------

extern "C" void kernel_launch(void* const* d_in, const int* in_sizes, int n_in,
                              void* d_out, int out_size, void* d_ws, size_t ws_size,
                              hipStream_t stream) {
  (void)in_sizes; (void)n_in; (void)out_size; (void)ws_size;
  const float* src  = (const float*)d_in[0];
  const float* tgt  = (const float*)d_in[1];
  const float* mask = (const float*)d_in[2];
  const float* Wq   = (const float*)d_in[3];
  const float* bq   = (const float*)d_in[4];
  const float* Wk   = (const float*)d_in[5];
  const float* bk   = (const float*)d_in[6];
  const float* Wv   = (const float*)d_in[7];
  const float* bv   = (const float*)d_in[8];
  const float* Wo   = (const float*)d_in[9];
  const float* bo   = (const float*)d_in[10];
  float* out = (float*)d_out;

  unsigned short* ws = (unsigned short*)d_ws;
  const size_t BSD = (size_t)BATCHC * SEQC * D_MODELC;     // 4,194,304
  const size_t WT  = (size_t)N_HEADC * D_QKVC * D_MODELC;  // 1,048,576
  unsigned short* tgt_bf = ws;  ws += BSD;
  unsigned short* src_bf = ws;  ws += BSD;
  unsigned short* WqT    = ws;  ws += WT;
  unsigned short* WkT    = ws;  ws += WT;
  unsigned short* WvT    = ws;  ws += WT;
  unsigned short* wo_bf  = ws;  ws += (size_t)D_MODELC * D_MODELC;
  unsigned short* q_bf   = ws;  ws += BSD;
  unsigned short* k_bf   = ws;  ws += BSD;
  unsigned short* vT_bf  = ws;  ws += BSD;
  unsigned short* ctx_bf = ws;  ws += BSD;

  const int TPB = 256;
  k_cast_bf16<<<2048, TPB, 0, stream>>>(tgt, tgt_bf, (int)BSD);
  k_cast_bf16<<<2048, TPB, 0, stream>>>(src, src_bf, (int)BSD);
  k_cast_bf16<<<1024, TPB, 0, stream>>>(Wo, wo_bf, D_MODELC * D_MODELC);
  k_cast_wT<<<(int)(WT / TPB), TPB, 0, stream>>>(Wq, WqT);
  k_cast_wT<<<(int)(WT / TPB), TPB, 0, stream>>>(Wk, WkT);
  k_cast_wT<<<(int)(WT / TPB), TPB, 0, stream>>>(Wv, WvT);

  k_qkv<<<dim3(BATCHC * SEQC / 128, N_HEADC, 3), 256, 0, stream>>>(
      tgt_bf, src_bf, WqT, WkT, WvT, bq, bk, bv, q_bf, k_bf, vT_bf);

  k_attn<<<dim3(SEQC / 128, N_HEADC, BATCHC), 256, 0, stream>>>(
      q_bf, k_bf, vT_bf, mask, ctx_bf);

  k_oproj<<<dim3(BATCHC * SEQC / 128, D_MODELC / 64), 256, 0, stream>>>(
      ctx_bf, wo_bf, bo, out);
}
